// CausalSelfAttention_6279242187027
// MI455X (gfx1250) — matrix operand order repaired and checked by reading, not compiled, not run on hardware
//
#include <hip/hip_runtime.h>
#include <math.h>
#include <stdint.h>

typedef unsigned short u16;

// ---------------- problem constants ----------------
constexpr int B_   = 4;
constexpr int T_   = 2048;
constexpr int C_   = 2048;
constexpr int NH_  = 16;
constexpr int HD_  = 128;   // head dim
constexpr int KVD_ = 512;   // C / N_REP (k or v width)
constexpr int KV2_ = 1024;  // 2 * KVD_
constexpr int MROWS = B_ * T_;  // 8192

// ---------------- types ----------------
typedef __attribute__((ext_vector_type(16))) __bf16 v16bf;
typedef __attribute__((ext_vector_type(8)))  float  v8f;

// ---------------- helpers ----------------
__device__ inline u16 f2bfu(float f) {
  unsigned u = __builtin_bit_cast(unsigned, f);
  u += 0x7FFFu + ((u >> 16) & 1u);   // round-to-nearest-even
  return (u16)(u >> 16);
}
__device__ inline __bf16 us2bf(u16 u) { return __builtin_bit_cast(__bf16, u); }

// CDNA5 async copy: 16 bytes global -> LDS, tracked by ASYNCcnt.
__device__ inline void async16(void* lds, const void* gptr) {
  unsigned l = (unsigned)(size_t)lds;   // low 32 bits of generic ptr == LDS offset
  asm volatile("global_load_async_to_lds_b128 %0, %1, off"
               :: "v"(l), "v"(gptr) : "memory");
}
__device__ inline void wait_async() {
  asm volatile("s_wait_asynccnt 0" ::: "memory");
}

// A-fragment (16x32 bf16), src = tile origin (row-major), ld = row stride.
// Lane l: M=l%16, half=l/16. VGPR r: K = (r<4?0:16) + 8*half + 2*(r%4) (+1).
__device__ inline v16bf load_a(const u16* src, int ld) {
  const int lane = threadIdx.x & 31;
  const int m = lane & 15, hf = lane >> 4;
  v16bf a;
#pragma unroll
  for (int r = 0; r < 8; ++r) {
    int kb = ((r & 4) << 2) + hf * 8 + (r & 3) * 2;
    a[2 * r]     = us2bf(src[m * ld + kb]);
    a[2 * r + 1] = us2bf(src[m * ld + kb + 1]);
  }
  return a;
}

// B-fragment (32x16) from N-MAJOR storage: element (k,n) at src[n*ld + k].
// Lane l: N=l%16, half=l/16. VGPR r: K = (r<4?0:16) + 8*half + 2*(r%4) (+1).
__device__ inline v16bf load_bt(const u16* src, int ld) {
  const int lane = threadIdx.x & 31;
  const int n = lane & 15, hf = lane >> 4;
  v16bf b;
#pragma unroll
  for (int r = 0; r < 8; ++r) {
    int k = ((r & 4) << 2) + hf * 8 + (r & 3) * 2;
    b[2 * r]     = us2bf(src[n * ld + k]);
    b[2 * r + 1] = us2bf(src[n * ld + k + 1]);
  }
  return b;
}

__device__ inline v8f wmma_bf16(v16bf a, v16bf b, v8f c) {
  return __builtin_amdgcn_wmma_f32_16x16x32_bf16(
      false, a, false, b, (short)0, c, false, false);
}

// ============================================================
// elementwise f32 -> bf16 (4 per thread)
// ============================================================
__global__ __launch_bounds__(256) void cvt_kernel(
    const float* __restrict__ src, u16* __restrict__ dst, int n)
{
  int i = (blockIdx.x * 256 + threadIdx.x) * 4;
  if (i < n) {
#pragma unroll
    for (int k = 0; k < 4; ++k) dst[i + k] = f2bfu(src[i + k]);
  }
}

// ============================================================
// transpose + cvt: dst[n*K + k] = bf16(src[k*N + n])  (tiled, coalesced)
// grid: (N/32, K/32), block: (32, 8)
// ============================================================
__global__ __launch_bounds__(256) void tcvt_kernel(
    const float* __restrict__ src, u16* __restrict__ dst, int K, int N)
{
  __shared__ float t[32][33];
  const int bx = blockIdx.x * 32;  // N
  const int by = blockIdx.y * 32;  // K
  for (int i = threadIdx.y; i < 32; i += 8)
    t[i][threadIdx.x] = src[(size_t)(by + i) * N + bx + threadIdx.x];
  __syncthreads();
  for (int i = threadIdx.y; i < 32; i += 8)
    dst[(size_t)(bx + i) * K + by + threadIdx.x] = f2bfu(t[threadIdx.x][i]);
}

// ============================================================
// GEMM: C[M,N] (f32) = A[M,K] (bf16) * WT[N,K]^T (bf16), all-WMMA.
// Block: 256 threads = 8 waves (4 x 2), tile 128(M) x 128(N), K-step 64,
// double-buffered async LDS staging.
// ============================================================
template <int N, int K>
__global__ __launch_bounds__(256) void gemm_kernel(
    const u16* __restrict__ A,   // [M][K] bf16
    const u16* __restrict__ WT,  // [N][K] bf16 (pre-transposed)
    float* __restrict__ Cout)    // [M][N] f32
{
  __shared__ u16 As[2][128][64];  // 16 KB x2
  __shared__ u16 Wt[2][128][64];  // 16 KB x2

  const int tid = threadIdx.x;
  const int lane = tid & 31, wave = tid >> 5;
  const int wm = wave >> 1, wn = wave & 1;   // 4x2 wave grid, wave tile 32x64
  const int hf = lane >> 4, ln = lane & 15;
  const int rowbase = blockIdx.y * 128;
  const int colbase = blockIdx.x * 128;

  auto stage = [&](int t, int db) {
    const int k0 = t * 64;
#pragma unroll
    for (int id = tid; id < 1024; id += 256) {       // A: 128 rows x 8 chunks
      int r = id >> 3, c = id & 7;
      async16(&As[db][r][c * 8], A + (size_t)(rowbase + r) * K + k0 + c * 8);
    }
#pragma unroll
    for (int id = tid; id < 1024; id += 256) {       // WT: 128 rows x 8 chunks
      int r = id >> 3, c = id & 7;
      async16(&Wt[db][r][c * 8], WT + (size_t)(colbase + r) * K + k0 + c * 8);
    }
  };

  v8f acc[2][4] = {};

  stage(0, 0);
  wait_async();
  __syncthreads();

  constexpr int NT = K / 64;
  for (int t = 0; t < NT; ++t) {
    const int db = t & 1;
    if (t + 1 < NT) stage(t + 1, db ^ 1);   // stream next tile during compute
#pragma unroll
    for (int kc = 0; kc < 2; ++kc) {
      v16bf a0 = load_a(&As[db][wm * 32][kc * 32], 64);
      v16bf a1 = load_a(&As[db][wm * 32 + 16][kc * 32], 64);
#pragma unroll
      for (int fn = 0; fn < 4; ++fn) {
        v16bf bb = load_bt(&Wt[db][wn * 64 + fn * 16][kc * 32], 64);
        acc[0][fn] = wmma_bf16(a0, bb, acc[0][fn]);
        acc[1][fn] = wmma_bf16(a1, bb, acc[1][fn]);
      }
    }
    wait_async();
    __syncthreads();
  }

#pragma unroll
  for (int fm = 0; fm < 2; ++fm)
#pragma unroll
    for (int fn = 0; fn < 4; ++fn)
#pragma unroll
      for (int r = 0; r < 8; ++r) {
        int row = rowbase + wm * 32 + fm * 16 + hf * 8 + r;
        int col = colbase + wn * 64 + fn * 16 + ln;
        Cout[(size_t)row * N + col] = acc[fm][fn][r];
      }
}

// ============================================================
// l2norm of q rows (C_) and k rows (first KVD_ cols of kv); v passthrough.
// The attention scale 1/sqrt(HD) is folded into q here.
// ============================================================
__device__ inline float block_sum256(float v, float* red) {
#pragma unroll
  for (int o = 16; o > 0; o >>= 1) v += __shfl_xor(v, o, 32);
  __syncthreads();
  if ((threadIdx.x & 31) == 0) red[threadIdx.x >> 5] = v;
  __syncthreads();
  float t = 0.f;
#pragma unroll
  for (int i = 0; i < 8; ++i) t += red[i];
  return t;
}

__global__ __launch_bounds__(256) void norm_kernel(
    const float* __restrict__ q, const float* __restrict__ kv,
    u16* __restrict__ qn, u16* __restrict__ kn, u16* __restrict__ vb)
{
  __shared__ float red[8];
  const int row = blockIdx.x;
  const float* qr  = q  + (size_t)row * C_;
  const float* kvr = kv + (size_t)row * KV2_;
  const float scale = 0.088388347648318447f;  // 1/sqrt(128), folded into q

  float s = 0.f;
  for (int i = threadIdx.x; i < C_; i += 256) { float t = qr[i]; s += t * t; }
  s = block_sum256(s, red);
  float invq = scale / (sqrtf(s) + 1e-12f);
  for (int i = threadIdx.x; i < C_; i += 256)
    qn[(size_t)row * C_ + i] = f2bfu(qr[i] * invq);

  float sk = 0.f;
  for (int i = threadIdx.x; i < KVD_; i += 256) { float t = kvr[i]; sk += t * t; }
  sk = block_sum256(sk, red);
  float invk = 1.0f / (sqrtf(sk) + 1e-12f);
  for (int i = threadIdx.x; i < KVD_; i += 256) {
    kn[(size_t)row * KVD_ + i] = f2bfu(kvr[i] * invk);
    vb[(size_t)row * KVD_ + i] = f2bfu(kvr[KVD_ + i]);
  }
}

// ============================================================
// Flash attention, causal, GQA (head h uses kv head g=h/4).
// Block = 128 threads (4 waves), 64 query rows (16/wave), 64-key tiles.
// Q A-frags preloaded to registers; masked path only on the diagonal tile.
// ============================================================
constexpr int BM = 64;
constexpr int BN = 64;
constexpr int VTS = 72;  // padded key-stride for Vt (bank-conflict dodge)

__global__ __launch_bounds__(128) void attn_kernel(
    const u16* __restrict__ qn,   // [B*T, C_] bf16 (pre-scaled by 1/sqrt(HD))
    const u16* __restrict__ kn,   // [B*T, KVD_] bf16
    const u16* __restrict__ vb,   // [B*T, KVD_] bf16
    u16* __restrict__ yb)         // [B*T, C_] bf16
{
  __shared__ u16 Qs[BM][HD_];      // 16 KB
  __shared__ u16 Ks[BN][HD_];      // 16 KB
  __shared__ u16 Vt[HD_][VTS];     // 18 KB, Vt[ch][key]
  __shared__ u16 Ps[4][16][BN];    //  8 KB

  const int tid = threadIdx.x;
  const int lane = tid & 31, wave = tid >> 5;
  const int hf = lane >> 4, ln = lane & 15;
  const int bh = blockIdx.y;
  const int b = bh >> 4, h = bh & 15;
  const int gidx = h >> 2;
  const int i0 = blockIdx.x * BM;
  const int m0 = wave * 16;

  // stage Q tile (async): 64 rows x 16 chunks of 16B
  for (int id = tid; id < BM * 16; id += 128) {
    int r = id >> 4, c = id & 15;
    async16(&Qs[r][c * 8],
            qn + (size_t)(b * T_ + i0 + r) * C_ + h * HD_ + c * 8);
  }
  wait_async();
  __syncthreads();

  // preload this wave's Q A-fragments (4 x v16bf = 32 VGPRs), Qs LDS is
  // not touched afterwards.
  v16bf Qa[4];
#pragma unroll
  for (int kc = 0; kc < 4; ++kc) Qa[kc] = load_a(&Qs[m0][kc * 32], HD_);

  v8f acc[8] = {};
  float mrow[8], lrow[8];
#pragma unroll
  for (int r = 0; r < 8; ++r) { mrow[r] = -INFINITY; lrow[r] = 0.f; }

  auto process_tile = [&](int j, bool masked) {
    __syncthreads();   // previous tile fully consumed
    // K tile: async copy, row-major (already the B^T layout for QK^T)
    for (int id = tid; id < BN * 16; id += 128) {
      int r = id >> 4, c = id & 15;
      async16(&Ks[r][c * 8],
              kn + (size_t)(b * T_ + j * BN + r) * KVD_ + gidx * HD_ + c * 8);
    }
    // V tile: manual transpose into Vt[ch][key] (coalesced global reads)
    for (int i = tid; i < BN * HD_; i += 128) {
      int r = i >> 7, c = i & (HD_ - 1);
      Vt[c][r] = vb[(size_t)(b * T_ + j * BN + r) * KVD_ + gidx * HD_ + c];
    }
    wait_async();
    __syncthreads();

    // ---- S = Q * K^T (16 x 64 per wave), K-dim = HD_ in 4 chunks ----
    v8f sfr[4] = {};
#pragma unroll
    for (int kc = 0; kc < 4; ++kc) {
#pragma unroll
      for (int nf = 0; nf < 4; ++nf) {
        v16bf bbf = load_bt(&Ks[nf * 16][kc * 32], HD_);
        sfr[nf] = wmma_bf16(Qa[kc], bbf, sfr[nf]);
      }
    }

    // ---- (diagonal only) causal mask; then row max ----
    float rmax[8];
#pragma unroll
    for (int r = 0; r < 8; ++r) rmax[r] = -INFINITY;
    const int rowb = i0 + m0 + hf * 8;
#pragma unroll
    for (int nf = 0; nf < 4; ++nf) {
      const int colrel = j * BN + nf * 16 + ln - rowb;  // col - row0
#pragma unroll
      for (int r = 0; r < 8; ++r) {
        float sv = sfr[nf][r];
        if (masked) sv = (colrel <= r) ? sv : -INFINITY;
        sfr[nf][r] = sv;
        rmax[r] = fmaxf(rmax[r], sv);
      }
    }
#pragma unroll
    for (int r = 0; r < 8; ++r) {   // 16-lane reduce (stays in row's half)
      float v = rmax[r];
      v = fmaxf(v, __shfl_xor(v, 1, 32));
      v = fmaxf(v, __shfl_xor(v, 2, 32));
      v = fmaxf(v, __shfl_xor(v, 4, 32));
      v = fmaxf(v, __shfl_xor(v, 8, 32));
      rmax[r] = v;
    }

    // ---- online softmax update ----
    float alpha[8], rsum[8];
#pragma unroll
    for (int r = 0; r < 8; ++r) {
      float mnew = fmaxf(mrow[r], rmax[r]);
      alpha[r] = __expf(mrow[r] - mnew);
      mrow[r] = mnew;
      rsum[r] = 0.f;
    }
#pragma unroll
    for (int nf = 0; nf < 4; ++nf) {
#pragma unroll
      for (int r = 0; r < 8; ++r) {
        float p = __expf(sfr[nf][r] - mrow[r]);
        rsum[r] += p;
        Ps[wave][hf * 8 + r][nf * 16 + ln] = f2bfu(p);  // D-frag -> A-frag via LDS
      }
    }
#pragma unroll
    for (int r = 0; r < 8; ++r) {
      float v = rsum[r];
      v += __shfl_xor(v, 1, 32);
      v += __shfl_xor(v, 2, 32);
      v += __shfl_xor(v, 4, 32);
      v += __shfl_xor(v, 8, 32);
      lrow[r] = lrow[r] * alpha[r] + v;
    }
#pragma unroll
    for (int of = 0; of < 8; ++of)
#pragma unroll
      for (int r = 0; r < 8; ++r)
        acc[of][r] *= alpha[r];

    __builtin_amdgcn_wave_barrier();  // per-wave LDS tile; DS in-order per wave

    // ---- O += P * V (16 x 128 per wave), K-dim = BN in 2 chunks ----
#pragma unroll
    for (int kc = 0; kc < 2; ++kc) {
      v16bf a = load_a(&Ps[wave][0][kc * 32], BN);
#pragma unroll
      for (int of = 0; of < 8; ++of) {
        v16bf bbf = load_bt(&Vt[of * 16][kc * 32], VTS);
        acc[of] = wmma_bf16(a, bbf, acc[of]);
      }
    }
    __builtin_amdgcn_wave_barrier();
  };

  // main loop: strictly-below-diagonal tiles need no mask
  for (int j = 0; j < blockIdx.x; ++j) process_tile(j, false);
  process_tile(blockIdx.x, true);   // diagonal tile

  // ---- normalize + store y (bf16, feeds proj GEMM) ----
#pragma unroll
  for (int of = 0; of < 8; ++of)
#pragma unroll
    for (int r = 0; r < 8; ++r) {
      int rowl = hf * 8 + r;
      yb[(size_t)(b * T_ + i0 + m0 + rowl) * C_ + h * HD_ + of * 16 + ln] =
          f2bfu(acc[of][r] / lrow[r]);
    }
}

// ============================================================
// launcher
// ============================================================
extern "C" void kernel_launch(void* const* d_in, const int* in_sizes, int n_in,
                              void* d_out, int out_size, void* d_ws, size_t ws_size,
                              hipStream_t stream)
{
  (void)in_sizes; (void)n_in; (void)out_size; (void)ws_size;
  const float* x     = (const float*)d_in[0];
  const float* Wq    = (const float*)d_in[1];
  const float* Wkv   = (const float*)d_in[2];
  const float* Wproj = (const float*)d_in[3];
  float* out = (float*)d_out;

  char* ws = (char*)d_ws;
  size_t off = 0;
  u16* xb   = (u16*)(ws + off); off += (size_t)MROWS * C_ * 2;       // 32 MB
  u16* wqT  = (u16*)(ws + off); off += (size_t)C_ * C_ * 2;          //  8 MB
  u16* wkvT = (u16*)(ws + off); off += (size_t)KV2_ * C_ * 2;        //  4 MB
  u16* wpT  = (u16*)(ws + off); off += (size_t)C_ * C_ * 2;          //  8 MB
  float* q  = (float*)(ws + off); off += (size_t)MROWS * C_ * 4;     // 64 MB
  float* kv = (float*)(ws + off); off += (size_t)MROWS * KV2_ * 4;   // 32 MB
  u16* qnb  = (u16*)(ws + off); off += (size_t)MROWS * C_ * 2;       // 32 MB
  u16* knb  = (u16*)(ws + off); off += (size_t)MROWS * KVD_ * 2;     //  8 MB
  u16* vbuf = (u16*)(ws + off); off += (size_t)MROWS * KVD_ * 2;     //  8 MB
  u16* ybuf = (u16*)q;  // q (f32) is dead after norm_kernel; reuse

  // 0) precision staging: x -> bf16; weights -> bf16 transposed
  cvt_kernel<<<dim3(MROWS * C_ / 1024), 256, 0, stream>>>(x, xb, MROWS * C_);
  tcvt_kernel<<<dim3(C_ / 32, C_ / 32), dim3(32, 8), 0, stream>>>(Wq, wqT, C_, C_);
  tcvt_kernel<<<dim3(KV2_ / 32, C_ / 32), dim3(32, 8), 0, stream>>>(Wkv, wkvT, C_, KV2_);
  tcvt_kernel<<<dim3(C_ / 32, C_ / 32), dim3(32, 8), 0, stream>>>(Wproj, wpT, C_, C_);

  // 1) q = x @ Wq ; 2) kv = x @ Wkv
  gemm_kernel<C_, C_><<<dim3(C_ / 128, MROWS / 128), 256, 0, stream>>>(xb, wqT, q);
  gemm_kernel<KV2_, C_><<<dim3(KV2_ / 128, MROWS / 128), 256, 0, stream>>>(xb, wkvT, kv);
  // 3) l2norm q,k (scale folded into q); bf16-ify q,k,v
  norm_kernel<<<dim3(MROWS), 256, 0, stream>>>(q, kv, qnb, knb, vbuf);
  // 4) flash attention -> y (bf16)
  attn_kernel<<<dim3(T_ / BM, B_ * NH_), 128, 0, stream>>>(qnb, knb, vbuf, ybuf);
  // 5) out = y @ Wproj
  gemm_kernel<C_, C_><<<dim3(C_ / 128, MROWS / 128), 256, 0, stream>>>(ybuf, wpT, out);
}
